// ProCata_50843822850557
// MI455X (gfx1250) — compile-verified, matrix operand order checked by reading, not run on hardware
//
#include <hip/hip_runtime.h>
#include <math.h>

typedef __attribute__((ext_vector_type(16))) _Float16 v16h;
typedef __attribute__((ext_vector_type(8)))  float    v8f;

#define HW 65536           // h*w = 256*256 tokens per image
#define CH 64              // channels == codebook dim
#define NC 512             // number of clusters
#define NB 4               // batch
#define TOK_PER_WAVE 16
#define WAVES 8
#define TOK_PER_BLOCK (TOK_PER_WAVE * WAVES)   // 128
#define TOK_STRIDE 68      // padded LDS row stride (floats): addr%64 = (4*row+c)%64
                           // -> conflict-free for float4 scatter stores AND frag reads

__device__ __forceinline__ void atomic_add_f32(float* p, float v) {
  (void)__hip_atomic_fetch_add(p, v, __ATOMIC_RELAXED, __HIP_MEMORY_SCOPE_AGENT);
}

// ---------------------------------------------------------------------------
// Kernel 1: zero accumulators, convert codebook f32 -> f16 once.
// ---------------------------------------------------------------------------
__global__ void vq_prep_kernel(const float* __restrict__ means,
                               _Float16* __restrict__ meansH,
                               float* __restrict__ sums,
                               int* __restrict__ bins) {
  int idx = blockIdx.x * blockDim.x + threadIdx.x;
  if (idx < NC * CH) {
    sums[idx]   = 0.0f;
    meansH[idx] = (_Float16)means[idx];
  }
  if (idx < NC) bins[idx] = 0;
}

// ---------------------------------------------------------------------------
// Kernel 2 (fused): WMMA similarity -> argmax bucket -> atomic scatter-add.
// Each wave owns a 16-token tile; block = 8 waves = 128 tokens; x read once.
// ---------------------------------------------------------------------------
__global__ void __launch_bounds__(256)
vq_assign_accum_kernel(const float* __restrict__ x,
                       const _Float16* __restrict__ meansH,
                       float* __restrict__ sums,
                       int* __restrict__ bins) {
  __shared__ float tokF[WAVES * TOK_PER_WAVE * TOK_STRIDE];  // ~34 KB

  const int lane = threadIdx.x & 31;
  const int wave = threadIdx.x >> 5;
  const int row  = lane & 15;       // token row within tile / cluster col for B
  const int hh   = lane >> 4;       // half-wave select (K-range per ISA layout)

  const int t0 = blockIdx.x * TOK_PER_BLOCK + wave * TOK_PER_WAVE;
  const int b  = t0 >> 16;          // HW == 65536 tokens per image
  const int l  = t0 & (HW - 1);     // tiles never straddle a batch boundary

  // ---- stage 16 tokens x 64 channels (f32), transposing channel-major x ----
  // float4 along the token dim: 8x global_load_b128 per lane = 512 B/wave-instr
  // on the only HBM-bound stream, then scatter-transpose into padded LDS.
  const float* xb = x + (size_t)b * (size_t)CH * HW + l;
  float* tw = &tokF[wave * (TOK_PER_WAVE * TOK_STRIDE)];
  {
    const int q  = lane & 3;        // which 4-token group (rows q*4 .. q*4+3)
    const int c0 = lane >> 2;       // channel offset within a pass (0..7)
    const float* xg = xb + q * 4;
#pragma unroll
    for (int p = 0; p < 8; ++p) {
      int c = p * 8 + c0;
      float4 f = *reinterpret_cast<const float4*>(xg + (size_t)c * HW);
      tw[(q * 4 + 0) * TOK_STRIDE + c] = f.x;
      tw[(q * 4 + 1) * TOK_STRIDE + c] = f.y;
      tw[(q * 4 + 2) * TOK_STRIDE + c] = f.z;
      tw[(q * 4 + 3) * TOK_STRIDE + c] = f.w;
    }
  }
  __syncthreads();

  // ---- A fragments: 16x32 f16, two k-steps cover K=64 ----
  // ISA 16-bit A layout: lanes 0-15 hold K {0..7,16..23}, lanes 16-31 {8..15,24..31}
  const float* trow = &tw[row * TOK_STRIDE];
  v16h a0, a1;
#pragma unroll
  for (int i = 0; i < 8; ++i) {
    a0[i]     = (_Float16)trow[hh * 8 + i];
    a0[8 + i] = (_Float16)trow[16 + hh * 8 + i];
    a1[i]     = (_Float16)trow[32 + hh * 8 + i];
    a1[8 + i] = (_Float16)trow[48 + hh * 8 + i];
  }

  float best[8];
  int   bidx[8];
#pragma unroll
  for (int r = 0; r < 8; ++r) { best[r] = -3.0e38f; bidx[r] = 0; }

  const int col = lane & 15;  // cluster column within the 16-wide N tile
#pragma unroll 2              // overlap next tile's loads/WMMA into the
                              // WMMA->VALU hazard gap of the current tile
  for (int nb = 0; nb < NC / 16; ++nb) {
    // B fragments: 32x16 f16, N across lanes, mirrored K layout.
    // Each 8-half run is 16B contiguous -> b128 loads; means f16 is L0-resident.
    const _Float16* mrow = meansH + (size_t)(nb * 16 + col) * CH;
    v16h b0, b1;
#pragma unroll
    for (int i = 0; i < 8; ++i) {
      b0[i]     = mrow[hh * 8 + i];
      b0[8 + i] = mrow[16 + hh * 8 + i];
      b1[i]     = mrow[32 + hh * 8 + i];
      b1[8 + i] = mrow[48 + hh * 8 + i];
    }
    v8f acc = {};
    acc = __builtin_amdgcn_wmma_f32_16x16x32_f16(false, a0, false, b0,
                                                 (short)0, acc, false, false);
    acc = __builtin_amdgcn_wmma_f32_16x16x32_f16(false, a1, false, b1,
                                                 (short)0, acc, false, false);
    // C layout: VGPR r -> (row r, lanes 0-15) / (row 8+r, lanes 16-31), N = lane&15
#pragma unroll
    for (int r = 0; r < 8; ++r) {
      float v = acc[r];
      if (v > best[r]) { best[r] = v; bidx[r] = nb * 16 + col; }
    }
  }

  // ---- argmax reduction across the 16 lanes of each half (first-max wins) ----
#pragma unroll
  for (int r = 0; r < 8; ++r) {
#pragma unroll
    for (int off = 8; off >= 1; off >>= 1) {
      float ov = __shfl_xor(best[r], off, 32);
      int   oi = __shfl_xor(bidx[r], off, 32);
      if (ov > best[r] || (ov == best[r] && oi < bidx[r])) {
        best[r] = ov; bidx[r] = oi;
      }
    }
  }

  // ---- scatter-add exact f32 token values into assigned centroids ----
  // token m's bucket lives (uniform) in lanes [(m>>3)*16 .. +15], slot r = m&7
#pragma unroll
  for (int m = 0; m < 16; ++m) {
    int bk = __shfl(bidx[m & 7], (m >> 3) << 4, 32);
    float v0 = tw[m * TOK_STRIDE + lane];        // channel = lane
    float v1 = tw[m * TOK_STRIDE + 32 + lane];   // channel = lane + 32
    atomic_add_f32(&sums[bk * CH + lane], v0);
    atomic_add_f32(&sums[bk * CH + 32 + lane], v1);
    if (lane == 0) atomicAdd(&bins[bk], 1);
  }
}

// ---------------------------------------------------------------------------
// Kernel 3: normalize per-cluster sums, empty-bucket guard, EMA update.
// One wave per cluster (wave32), 2 channels per lane.
// ---------------------------------------------------------------------------
__global__ void vq_finalize_kernel(const float* __restrict__ means,
                                   const float* __restrict__ sums,
                                   const int* __restrict__ bins,
                                   float* __restrict__ out) {
  const int c    = blockIdx.x;
  const int lane = threadIdx.x;
  float s0 = sums[c * CH + lane];
  float s1 = sums[c * CH + 32 + lane];
  float sq = s0 * s0 + s1 * s1;
#pragma unroll
  for (int off = 16; off >= 1; off >>= 1)
    sq += __shfl_xor(sq, off, 32);
  float denom = fmaxf(sqrtf(sq), 1e-12f);
  int   cnt = bins[c];
  float m0 = means[c * CH + lane];
  float m1 = means[c * CH + 32 + lane];
  float n0 = (cnt == 0) ? m0 : s0 / denom;
  float n1 = (cnt == 0) ? m1 : s1 / denom;
  out[c * CH + lane]      = 0.999f * m0 + 0.001f * n0;
  out[c * CH + 32 + lane] = 0.999f * m1 + 0.001f * n1;
}

// ---------------------------------------------------------------------------
extern "C" void kernel_launch(void* const* d_in, const int* in_sizes, int n_in,
                              void* d_out, int out_size, void* d_ws, size_t ws_size,
                              hipStream_t stream) {
  const float* x     = (const float*)d_in[0];   // (4, 64, 256, 256) f32
  const float* means = (const float*)d_in[1];   // (512, 64) f32

  // workspace layout: sums f32[512*64] | bins i32[512] | meansH f16[512*64]
  float*    sums   = (float*)d_ws;
  int*      bins   = (int*)((char*)d_ws + (size_t)NC * CH * sizeof(float));
  _Float16* meansH = (_Float16*)((char*)d_ws + (size_t)NC * CH * sizeof(float)
                                             + (size_t)NC * sizeof(int));

  vq_prep_kernel<<<(NC * CH + 255) / 256, 256, 0, stream>>>(means, meansH, sums, bins);

  const int n_tokens = NB * HW;  // 262144
  vq_assign_accum_kernel<<<n_tokens / TOK_PER_BLOCK, 256, 0, stream>>>(
      x, meansH, sums, bins);

  vq_finalize_kernel<<<NC, 32, 0, stream>>>(means, sums, bins, (float*)d_out);
}
